// LSTM_3006477107375
// MI455X (gfx1250) — matrix operand order repaired and checked by reading, not compiled, not run on hardware
//
#include <hip/hip_runtime.h>

// ---------------------------------------------------------------------------
// LSTM (SEQ=128, IN_DIM=1, H=1024, NCLS=10, B=1024) for gfx1250 (MI455X).
//  * Batch rows independent through the recurrence -> 64 blocks x 16 rows,
//    no inter-block sync; h (LDS f16) and c (VGPRs) carried across 128 steps.
//  * Per-step matmul h[16,1024] @ Wh^T[1024,4096] via v_wmma_f32_16x16x32_f16,
//    f32 accumulate. Wh converted once to f16 [4096,1024] row-major in ws
//    (8 MB, L2-resident: 192 MB L2) so B-tiles are contiguous b128 loads.
//  * 512 threads = 16 waves/block (4 waves/SIMD32 for latency hiding);
//    each wave owns 4 j-tiles (16 hidden cols each), c/h_new in v8f registers
//    matching the WMMA C/D layout.
//  * K-loop unrolled x2 so independent loads/address math fill the WMMA
//    WAR-hazard co-exec slots (replaces v_nop padding).
//  * Branch-free activations (v_exp/v_rcp, hw v_tanh_f32 when available);
//    keeps EXEC all-1s through the WMMA region.
// ---------------------------------------------------------------------------

typedef _Float16 v16h __attribute__((ext_vector_type(16)));
typedef _Float16 v8h  __attribute__((ext_vector_type(8)));
typedef float    v8f  __attribute__((ext_vector_type(8)));

#define SEQ   128
#define HDIM  1024
#define NCLS  10
#define ROWS  16      // batch rows per block (= WMMA M)
#define NTHR  512     // 16 waves
#define QT    4       // j-tiles per wave (64 tiles / 16 waves)

union F16x16 { v16h v; v8h h[2]; };

__global__ __launch_bounds__(256) void lstm_prep_kernel(
    const float* __restrict__ wg, const float* __restrict__ wi,
    const float* __restrict__ wf, const float* __restrict__ wo,
    _Float16* __restrict__ whf)
{
    const int idx = blockIdx.x * 256 + threadIdx.x;       // 0 .. 4M-1
    const int src = idx >> 20;
    const int off = idx & ((1 << 20) - 1);
    float v;
    if      (src == 0) v = wg[off];
    else if (src == 1) v = wi[off];
    else if (src == 2) v = wf[off];
    else               v = wo[off];
    whf[idx] = (_Float16)v;
}

// Branch-free activations. Inputs here are small (weights std=0.01), but these
// are also safe at extremes: exp->inf / rcp->0 saturate correctly, no NaN.
__device__ __forceinline__ float sigmoid_f(float x) {
    return __builtin_amdgcn_rcpf(1.0f + __expf(-x));
}
__device__ __forceinline__ float tanh_f(float x) {
#if __has_builtin(__builtin_amdgcn_tanhf)
    return __builtin_amdgcn_tanhf(x);          // CDNA5 v_tanh_f32
#else
    return 1.0f - 2.0f * __builtin_amdgcn_rcpf(1.0f + __expf(2.0f * x));
#endif
}

__global__ __launch_bounds__(NTHR) void lstm_step_kernel(
    const float* __restrict__ x,                          // [1024,128]
    const float* __restrict__ wxg, const float* __restrict__ wxi,
    const float* __restrict__ wxf, const float* __restrict__ wxo, // [1024]
    const float* __restrict__ bg,  const float* __restrict__ bi,
    const float* __restrict__ bf,  const float* __restrict__ bo,  // [1024]
    const float* __restrict__ wph, const float* __restrict__ bp,  // [10,1024],[10]
    const _Float16* __restrict__ whf,                     // [4096,1024] f16
    float* __restrict__ out)                              // [1024,10]
{
    __shared__ __align__(16) _Float16 hbuf[ROWS * HDIM];  // 32 KB current h (f16)
    __shared__ float xs[ROWS * SEQ];                      // 8 KB x slab

    const int tid  = threadIdx.x;
    const int wave = tid >> 5;
    const int lane = tid & 31;
    const int nl   = lane & 15;   // B col (N) within tile / A row (M)
    const int hi   = lane >> 4;   // lane-group: K sub-range select
    const int blk  = blockIdx.x;

    for (int i = tid; i < ROWS * HDIM; i += NTHR) hbuf[i] = (_Float16)0.0f;
    {
        const float* xsrc = x + (size_t)blk * ROWS * SEQ; // 16 contiguous rows
        for (int i = tid; i < ROWS * SEQ; i += NTHR) xs[i] = xsrc[i];
    }
    __syncthreads();

    v8f cst[QT];  // cell state, C/D layout
    v8f hn[QT];   // h_new staging (held across first barrier)
    #pragma unroll
    for (int q = 0; q < QT; ++q) { v8f z = {}; cst[q] = z; hn[q] = z; }

    #pragma unroll 1
    for (int t = 0; t < SEQ; ++t) {
        #pragma unroll
        for (int q = 0; q < QT; ++q) {
            const int jt = wave * QT + q;         // j-tile (0..63)
            const int j  = jt * 16 + nl;          // hidden column for this lane

            const _Float16* ap = hbuf + (nl * HDIM + 16 * hi);
            const _Float16* b0 = whf + ((size_t)(0 * HDIM + j) * HDIM + 16 * hi);
            const _Float16* b1 = whf + ((size_t)(1 * HDIM + j) * HDIM + 16 * hi);
            const _Float16* b2 = whf + ((size_t)(2 * HDIM + j) * HDIM + 16 * hi);
            const _Float16* b3 = whf + ((size_t)(3 * HDIM + j) * HDIM + 16 * hi);

            v8f acc0 = {}, acc1 = {}, acc2 = {}, acc3 = {};

            #pragma unroll 2
            for (int k0 = 0; k0 < HDIM; k0 += 32) {
                F16x16 a;
                a.h[0] = *(const v8h*)(ap + k0);        // K = k0 + 16*hi  .. +7
                a.h[1] = *(const v8h*)(ap + k0 + 8);    // K = k0+16*hi+8 .. +7
                F16x16 b;
                b.h[0] = *(const v8h*)(b0 + k0);
                b.h[1] = *(const v8h*)(b0 + k0 + 8);
                acc0 = __builtin_amdgcn_wmma_f32_16x16x32_f16(
                           false, a.v, false, b.v, (short)0, acc0, false, false);
                b.h[0] = *(const v8h*)(b1 + k0);
                b.h[1] = *(const v8h*)(b1 + k0 + 8);
                acc1 = __builtin_amdgcn_wmma_f32_16x16x32_f16(
                           false, a.v, false, b.v, (short)0, acc1, false, false);
                b.h[0] = *(const v8h*)(b2 + k0);
                b.h[1] = *(const v8h*)(b2 + k0 + 8);
                acc2 = __builtin_amdgcn_wmma_f32_16x16x32_f16(
                           false, a.v, false, b.v, (short)0, acc2, false, false);
                b.h[0] = *(const v8h*)(b3 + k0);
                b.h[1] = *(const v8h*)(b3 + k0 + 8);
                acc3 = __builtin_amdgcn_wmma_f32_16x16x32_f16(
                           false, a.v, false, b.v, (short)0, acc3, false, false);
            }

            const float wg_s = wxg[j], wi_s = wxi[j], wf_s = wxf[j], wo_s = wxo[j];
            const float bg_s = bg[j],  bi_s = bi[j],  bf_s = bf[j],  bo_s = bo[j];

            #pragma unroll
            for (int r = 0; r < 8; ++r) {
                const int m  = r + 8 * hi;                // C/D layout row
                const float xv = xs[m * SEQ + t];
                const float pg = acc0[r] + bg_s + xv * wg_s;
                const float pi = acc1[r] + bi_s + xv * wi_s;
                const float pf = acc2[r] + bf_s + xv * wf_s;
                const float po = acc3[r] + bo_s + xv * wo_s;
                const float g  = tanh_f(pg);
                const float ig = sigmoid_f(pi);
                const float fg = sigmoid_f(pf);
                const float og = sigmoid_f(po);
                const float cn = g * ig + cst[q][r] * fg;
                cst[q][r] = cn;
                hn[q][r]  = tanh_f(cn) * og;
            }
        }

        __syncthreads();   // all waves finished reading hbuf for this step

        #pragma unroll
        for (int q = 0; q < QT; ++q) {
            const int j = (wave * QT + q) * 16 + nl;
            #pragma unroll
            for (int r = 0; r < 8; ++r) {
                const int m = r + 8 * hi;
                hbuf[m * HDIM + j] = (_Float16)hn[q][r];
            }
        }

        __syncthreads();   // new h visible before next step's reads
    }

    // ---- final projection: logits = h_last @ w_ph^T + bias_p (tiny) ----
    if (tid < ROWS * NCLS) {
        const int bl  = tid / NCLS;
        const int cls = tid % NCLS;
        float s = bp[cls];
        const float*    wp = wph + cls * HDIM;
        const _Float16* hp = hbuf + bl * HDIM;
        for (int k = 0; k < HDIM; ++k) s += (float)hp[k] * wp[k];
        out[((size_t)blk * ROWS + bl) * NCLS + cls] = s;
    }
}

extern "C" void kernel_launch(void* const* d_in, const int* in_sizes, int n_in,
                              void* d_out, int out_size, void* d_ws, size_t ws_size,
                              hipStream_t stream) {
    const float* x   = (const float*)d_in[0];
    const float* wgx = (const float*)d_in[1];
    const float* wgh = (const float*)d_in[2];
    const float* wix = (const float*)d_in[3];
    const float* wih = (const float*)d_in[4];
    const float* wfx = (const float*)d_in[5];
    const float* wfh = (const float*)d_in[6];
    const float* wox = (const float*)d_in[7];
    const float* woh = (const float*)d_in[8];
    const float* wph = (const float*)d_in[9];
    const float* bg  = (const float*)d_in[10];
    const float* bi  = (const float*)d_in[11];
    const float* bf  = (const float*)d_in[12];
    const float* bo  = (const float*)d_in[13];
    const float* bp  = (const float*)d_in[14];

    _Float16* whf = (_Float16*)d_ws;            // 4096*1024 f16 = 8 MB
    float*    out = (float*)d_out;

    lstm_prep_kernel<<<(4 * HDIM * HDIM) / 256, 256, 0, stream>>>(
        wgh, wih, wfh, woh, whf);

    lstm_step_kernel<<<1024 / ROWS, NTHR, 0, stream>>>(
        x, wgx, wix, wfx, wox, bg, bi, bf, bo, wph, bp, whf, out);
}
